// GAT_NET_69707319214252
// MI455X (gfx1250) — compile-verified, matrix operand order checked by reading, not run on hardware
//
#include <hip/hip_runtime.h>
#include <hip/hip_bf16.h>

#define HEADS 4
#define OUTC  64
#define INC   256   // IN_CH == HEADS*OUT_CH == 256

typedef __attribute__((ext_vector_type(16))) _Float16 v16h;
typedef __attribute__((ext_vector_type(4)))  _Float16 v4h;
typedef __attribute__((ext_vector_type(8)))  float    v8f;

static __device__ __forceinline__ int fkey(float f) {
  int i = __float_as_int(f);
  return (i >= 0) ? i : (i ^ 0x7fffffff);   // order-preserving int key for float
}
static __device__ __forceinline__ float funkey(int k) {
  return __int_as_float((k >= 0) ? k : (k ^ 0x7fffffff));
}
static __device__ __forceinline__ float lrelu(float a) {
  return a > 0.0f ? a : 0.2f * a;
}

// ---------------- init: zero accumulator, -inf max keys, zero denom ----------
__global__ void k_init(float4* __restrict__ out4, int n4,
                       int* __restrict__ amaxk, float* __restrict__ denom, int nh) {
  int i = blockIdx.x * 256 + threadIdx.x;
  if (i < n4) out4[i] = make_float4(0.f, 0.f, 0.f, 0.f);
  if (i < nh) { amaxk[i] = (int)0x80000000; denom[i] = 0.f; }
}

// ---------------- convert x (f32) -> xh (f16) --------------------------------
__global__ void k_cvt_x(const float4* __restrict__ x4, v4h* __restrict__ xh4, int n4) {
  int i = blockIdx.x * 256 + threadIdx.x;
  if (i >= n4) return;
  float4 v = x4[i];
  v4h o;
  o.x = (_Float16)v.x; o.y = (_Float16)v.y; o.z = (_Float16)v.z; o.w = (_Float16)v.w;
  xh4[i] = o;
}

// ---------------- pre-swizzle W (f32 row-major [K=256][N=256]) into f16 B-layout
// B layout per wmma_f32_16x16x32_f16 (32x16, 16-bit): lane<16 -> N=lane,
//   elems 0..7 = K k0+0..7, elems 8..15 = K k0+16..23; lanes 16..31 mirrored (+8).
// bsw[((ct*8 + ks)*32 + lane)*16 + j]
__global__ void k_prep_w(const float* __restrict__ W, _Float16* __restrict__ bsw) {
  int t = blockIdx.x * 256 + threadIdx.x;
  if (t >= 16 * 8 * 32) return;
  int lane = t & 31;
  int ks   = (t >> 5) & 7;
  int ct   = t >> 8;
  int hs   = lane >> 4;
  int n    = lane & 15;
  int k0   = ks * 32;
  _Float16* dst = bsw + (size_t)t * 16;
#pragma unroll
  for (int j = 0; j < 16; ++j) {
    int kk = k0 + hs * 8 + ((j < 8) ? j : (16 + (j - 8)));
    dst[j] = (_Float16)W[(size_t)kk * INC + ct * 16 + n];
  }
}

// ---------------- GEMM: h = x @ W via v_wmma_f32_16x16x32_f16 ----------------
// one wave per 16x16 output tile; ntiles_row = N/16 (N%16==0), 16 col tiles.
__global__ void k_gemm(const _Float16* __restrict__ xh, const _Float16* __restrict__ bsw,
                       float* __restrict__ hout, int ntiles_row) {
  int gwave = (blockIdx.x * blockDim.x + threadIdx.x) >> 5;
  int lane  = threadIdx.x & 31;
  if (gwave >= ntiles_row * 16) return;
  int rt = gwave >> 4;      // row tile
  int ct = gwave & 15;      // col tile
  int m  = lane & 15;
  int hs = lane >> 4;

  const _Float16* arow = xh + (size_t)(rt * 16 + m) * INC;
  v8f c = {};
#pragma unroll
  for (int ks = 0; ks < 8; ++ks) {
    int k0 = ks * 32;
    v16h a, b;
    // A: 16-bit A 16x32 layout -> two contiguous 16B chunks per lane
    ((float4*)&a)[0] = *(const float4*)(arow + k0 + hs * 8);
    ((float4*)&a)[1] = *(const float4*)(arow + k0 + 16 + hs * 8);
    const _Float16* bp = bsw + ((size_t)((ct * 8 + ks) * 32 + lane)) * 16;
    ((float4*)&b)[0] = ((const float4*)bp)[0];
    ((float4*)&b)[1] = ((const float4*)bp)[1];
    c = __builtin_amdgcn_wmma_f32_16x16x32_f16(false, a, false, b, (short)0, c,
                                               false, false);
  }
  // C/D layout: VGPR v -> M = hs*8 + v, N = lane&15
  float* ob = hout + (size_t)(rt * 16 + hs * 8) * INC + ct * 16 + m;
#pragma unroll
  for (int v = 0; v < 8; ++v) ob[(size_t)v * INC] = c[v];
}

// ---------------- per-(node,head) attention dots -----------------------------
__global__ void k_att(const float* __restrict__ h, const float* __restrict__ att_src,
                      const float* __restrict__ att_dst,
                      float* __restrict__ a_src, float* __restrict__ a_dst, int nh) {
  int t = blockIdx.x * 256 + threadIdx.x;  // node*4 + head
  if (t >= nh) return;
  int n  = t >> 2;
  int hd = t & 3;
  const float* hp = h + (size_t)n * INC + hd * OUTC;
  const float* as = att_src + hd * OUTC;
  const float* ad = att_dst + hd * OUTC;
  float s1 = 0.f, s2 = 0.f;
#pragma unroll 4
  for (int c = 0; c < OUTC; c += 4) {
    float4 hv = *(const float4*)(hp + c);
    float4 a1 = *(const float4*)(as + c);
    float4 a2 = *(const float4*)(ad + c);
    s1 += hv.x * a1.x + hv.y * a1.y + hv.z * a1.z + hv.w * a1.w;
    s2 += hv.x * a2.x + hv.y * a2.y + hv.z * a2.z + hv.w * a2.w;
  }
  a_src[t] = s1;
  a_dst[t] = s2;
}

// ---------------- segment max over dst (int-keyed float atomic max) ----------
__global__ void k_amax(const int* __restrict__ ei, int E, int N,
                       const float* __restrict__ a_src, const float* __restrict__ a_dst,
                       int* __restrict__ amaxk) {
  int e = blockIdx.x * 256 + threadIdx.x;
  if (e >= E + N) return;
  int s, d;
  if (e < E) { s = ei[e]; d = ei[E + e]; } else { s = d = e - E; }
  float4 as = *(const float4*)(a_src + (size_t)s * 4);
  float4 ad = *(const float4*)(a_dst + (size_t)d * 4);
  int* am = amaxk + (size_t)d * 4;
  atomicMax(am + 0, fkey(lrelu(as.x + ad.x)));
  atomicMax(am + 1, fkey(lrelu(as.y + ad.y)));
  atomicMax(am + 2, fkey(lrelu(as.z + ad.z)));
  atomicMax(am + 3, fkey(lrelu(as.w + ad.w)));
}

// ---------------- segment sum of exp(alpha - amax) ---------------------------
__global__ void k_denom(const int* __restrict__ ei, int E, int N,
                        const float* __restrict__ a_src, const float* __restrict__ a_dst,
                        const int* __restrict__ amaxk, float* __restrict__ denom) {
  int e = blockIdx.x * 256 + threadIdx.x;
  if (e >= E + N) return;
  int s, d;
  if (e < E) { s = ei[e]; d = ei[E + e]; } else { s = d = e - E; }
  float4 as = *(const float4*)(a_src + (size_t)s * 4);
  float4 ad = *(const float4*)(a_dst + (size_t)d * 4);
  const int* am = amaxk + (size_t)d * 4;
  float* dn = denom + (size_t)d * 4;
  atomicAdd(dn + 0, __expf(lrelu(as.x + ad.x) - funkey(am[0])));
  atomicAdd(dn + 1, __expf(lrelu(as.y + ad.y) - funkey(am[1])));
  atomicAdd(dn + 2, __expf(lrelu(as.z + ad.z) - funkey(am[2])));
  atomicAdd(dn + 3, __expf(lrelu(as.w + ad.w) - funkey(am[3])));
}

// ---------------- weighted scatter: out[dst] += coef * h[src] ----------------
// 64 threads per edge, 4 channels (float4) per thread.
__global__ void k_scatter(const int* __restrict__ ei, int E, int N,
                          const float* __restrict__ h,
                          const float* __restrict__ a_src, const float* __restrict__ a_dst,
                          const int* __restrict__ amaxk, const float* __restrict__ denom,
                          float* __restrict__ out) {
  long long t = (long long)blockIdx.x * 256 + threadIdx.x;
  int e = (int)(t >> 6);
  if (e >= E + N) return;
  int lt = (int)(t & 63);
  int c0 = lt * 4;          // 0..252
  int hd = c0 >> 6;         // head = c0/64
  int s, d;
  if (e < E) { s = ei[e]; d = ei[E + e]; } else { s = d = e - E; }
  float alpha = lrelu(a_src[(size_t)s * 4 + hd] + a_dst[(size_t)d * 4 + hd]);
  float ex    = __expf(alpha - funkey(amaxk[(size_t)d * 4 + hd]));
  float coef  = ex / (denom[(size_t)d * 4 + hd] + 1e-16f);
  float4 hv = *(const float4*)(h + (size_t)s * INC + c0);
  float* op = out + (size_t)d * INC + c0;
  atomicAdd(op + 0, coef * hv.x);
  atomicAdd(op + 1, coef * hv.y);
  atomicAdd(op + 2, coef * hv.z);
  atomicAdd(op + 3, coef * hv.w);
}

// ---------------- epilogue: out = relu(out + bias), in place -----------------
__global__ void k_final(float4* __restrict__ out4, const float* __restrict__ bias, int n4) {
  int i = blockIdx.x * 256 + threadIdx.x;
  if (i >= n4) return;
  float4 v = out4[i];
  int cb = (i * 4) & (INC - 1);
  v.x = fmaxf(v.x + bias[cb + 0], 0.f);
  v.y = fmaxf(v.y + bias[cb + 1], 0.f);
  v.z = fmaxf(v.z + bias[cb + 2], 0.f);
  v.w = fmaxf(v.w + bias[cb + 3], 0.f);
  out4[i] = v;
}

static inline size_t alignup(size_t x) { return (x + 255) & ~(size_t)255; }

extern "C" void kernel_launch(void* const* d_in, const int* in_sizes, int n_in,
                              void* d_out, int out_size, void* d_ws, size_t ws_size,
                              hipStream_t stream) {
  const float* x       = (const float*)d_in[0];
  const int*   ei      = (const int*)d_in[1];
  const float* W       = (const float*)d_in[2];
  const float* att_src = (const float*)d_in[3];
  const float* att_dst = (const float*)d_in[4];
  const float* bias    = (const float*)d_in[5];

  const int N = in_sizes[0] / INC;   // 50000
  const int E = in_sizes[1] / 2;     // 800000
  const int NH = N * HEADS;

  // workspace layout
  char* ws = (char*)d_ws;
  size_t off = 0;
  _Float16* xh  = (_Float16*)(ws + off); off = alignup(off + (size_t)N * INC * 2);
  _Float16* bsw = (_Float16*)(ws + off); off = alignup(off + (size_t)16 * 8 * 32 * 16 * 2);
  float*    h   = (float*)(ws + off);    off = alignup(off + (size_t)N * INC * 4);
  float*    asr = (float*)(ws + off);    off = alignup(off + (size_t)NH * 4);
  float*    ads = (float*)(ws + off);    off = alignup(off + (size_t)NH * 4);
  int*      amx = (int*)(ws + off);      off = alignup(off + (size_t)NH * 4);
  float*    dnm = (float*)(ws + off);    off = alignup(off + (size_t)NH * 4);
  if (off > ws_size) return;  // insufficient scratch

  float* out = (float*)d_out;

  const int n4   = N * INC / 4;                       // float4 count of out / x
  const int grid4 = (n4 + 255) / 256;

  // 1) init accumulator + softmax state
  k_init<<<grid4, 256, 0, stream>>>((float4*)out, n4, amx, dnm, NH);
  // 2) x -> f16
  k_cvt_x<<<grid4, 256, 0, stream>>>((const float4*)x, (v4h*)xh, n4);
  // 3) W -> swizzled f16 B tiles
  k_prep_w<<<(16 * 8 * 32 + 255) / 256, 256, 0, stream>>>(W, bsw);
  // 4) h = x @ W  (WMMA f16->f32)
  {
    int ntr = N / 16;                  // 3125 (N is a multiple of 16)
    int waves = ntr * 16;
    int blocks = (waves * 32 + 255) / 256;
    k_gemm<<<blocks, 256, 0, stream>>>(xh, bsw, h, ntr);
  }
  // 5) attention dots
  k_att<<<(NH + 255) / 256, 256, 0, stream>>>(h, att_src, att_dst, asr, ads, NH);
  // 6) segment max
  k_amax<<<(E + N + 255) / 256, 256, 0, stream>>>(ei, E, N, asr, ads, amx);
  // 7) segment sum of exp
  k_denom<<<(E + N + 255) / 256, 256, 0, stream>>>(ei, E, N, asr, ads, amx, dnm);
  // 8) weighted scatter
  {
    long long threads = (long long)(E + N) * 64;
    int blocks = (int)((threads + 255) / 256);
    k_scatter<<<blocks, 256, 0, stream>>>(ei, E, N, h, asr, ads, amx, dnm, out);
  }
  // 9) bias + relu
  k_final<<<grid4, 256, 0, stream>>>((float4*)out, bias, n4);
}
